// CustomModel_35407710388587
// MI455X (gfx1250) — matrix operand order repaired and checked by reading, not compiled, not run on hardware
//
#include <hip/hip_runtime.h>
#include <math.h>

#define NNODES 100000
#define NEDGES 2048
#define LTOK   8
#define HID    64
#define DIN    512
#define DGNN   128
#define VOCAB  32000
#define FFND   256

typedef __attribute__((ext_vector_type(8)))  float  v8f;
typedef __attribute__((ext_vector_type(16))) __bf16 v16bf;

struct BF16x16Bits { uint4 lo, hi; };

__device__ __forceinline__ __bf16 f2bf(float f) {
  unsigned u = __builtin_bit_cast(unsigned, f);
  u += 0x7FFFu + ((u >> 16) & 1u);
  unsigned short s = (unsigned short)(u >> 16);
  return __builtin_bit_cast(__bf16, s);
}
__device__ __forceinline__ float bf2f(__bf16 h) {
  unsigned short s = __builtin_bit_cast(unsigned short, h);
  unsigned u = ((unsigned)s) << 16;
  return __builtin_bit_cast(float, u);
}
__device__ __forceinline__ v16bf load_b16(const __bf16* p) {
  BF16x16Bits t;
  t.lo = *(const uint4*)p;
  t.hi = *(const uint4*)(p + 8);
  return __builtin_bit_cast(v16bf, t);
}
__device__ __forceinline__ v16bf pack2(uint4 lo, uint4 hi) {
  BF16x16Bits t; t.lo = lo; t.hi = hi;
  return __builtin_bit_cast(v16bf, t);
}

// ---------------- packing kernels ----------------

__global__ void k_f32_to_bf16(const float* __restrict__ src, __bf16* __restrict__ dst, int n) {
  int t = blockIdx.x * blockDim.x + threadIdx.x;
  if (t < n) dst[t] = f2bf(src[t]);
}

// Pack row-major W[K][N] (K%32==0, N%16==0) into WMMA-B tiles:
// packed[(kt*(N/16)+nt)*512 + lane*16 + j] = W[kt*32 + 16*(j>>3) + 8*(lane>>4) + (j&7)][nt*16 + (lane&15)]
__global__ void k_pack_wmma_b(const float* __restrict__ W, __bf16* __restrict__ dst, int K, int N) {
  int t = blockIdx.x * blockDim.x + threadIdx.x;
  if (t >= K * N) return;
  int j    = t & 15;
  int lane = (t >> 4) & 31;
  int tile = t >> 9;
  int ntiles = N >> 4;
  int nt = tile % ntiles;
  int kt = tile / ntiles;
  int hi = lane >> 4;
  int k = kt * 32 + (j & 8) * 2 + hi * 8 + (j & 7);
  int n = nt * 16 + (lane & 15);
  dst[t] = f2bf(W[(size_t)k * N + n]);
}

// ---------------- GNN ----------------

// zero agg rows for every edge endpoint (src and dst); duplicates all write 0.
__global__ void k_zero_agg(const int* __restrict__ edge_index, float* __restrict__ agg) {
  int r = blockIdx.x;                       // 0..2*NEDGES-1 ; even->src, odd->dst
  int node = edge_index[(r & 1) * NEDGES + (r >> 1)];
  agg[(size_t)node * DGNN + threadIdx.x] = 0.f;
}

// msg = (node_emb[src]+edge_emb) @ W_nbr ; atomic scatter-add into agg[dst]
// N=128 split into two passes of 4 accumulators to avoid VGPR spills.
__global__ void k_msg_scatter(const int* __restrict__ node_tokens,
                              const int* __restrict__ edge_tokens,
                              const int* __restrict__ edge_index,
                              const __bf16* __restrict__ embT,
                              const __bf16* __restrict__ WnbrP,
                              float* __restrict__ agg) {
  __shared__ int sdst[16];
  int l = threadIdx.x;
  int eb = blockIdx.x * 16;
  if (l < 16) sdst[l] = edge_index[NEDGES + eb + l];
  __syncthreads();
  int m = l & 15, hi = l >> 4;
  int e = eb + m;
  int srcn = edge_index[e];
  int ts[LTOK], te[LTOK];
#pragma unroll
  for (int s = 0; s < LTOK; ++s) {
    ts[s] = node_tokens[srcn * LTOK + s];
    te[s] = edge_tokens[e * LTOK + s];
  }
  for (int half = 0; half < 2; ++half) {
    v8f c[4] = {};
    for (int kt = 0; kt < DIN / 32; ++kt) {
      int k0 = kt * 32 + hi * 8;
      int k1 = k0 + 16;
      uint4 nlo = *(const uint4*)(embT + (size_t)ts[k0 >> 6] * HID + (k0 & 63));
      uint4 nhi = *(const uint4*)(embT + (size_t)ts[k1 >> 6] * HID + (k1 & 63));
      uint4 elo = *(const uint4*)(embT + (size_t)te[k0 >> 6] * HID + (k0 & 63));
      uint4 ehi = *(const uint4*)(embT + (size_t)te[k1 >> 6] * HID + (k1 & 63));
      v16bf an = pack2(nlo, nhi);
      v16bf ae = pack2(elo, ehi);
      v16bf a;
#pragma unroll
      for (int j = 0; j < 16; ++j) a[j] = f2bf(bf2f(an[j]) + bf2f(ae[j]));
#pragma unroll
      for (int n2 = 0; n2 < 4; ++n2) {
        int nt = half * 4 + n2;
        v16bf b = load_b16(WnbrP + ((size_t)(kt * (DGNN / 16) + nt) << 9) + l * 16);
        c[n2] = __builtin_amdgcn_wmma_f32_16x16x32_bf16(false, a, false, b, (short)0, c[n2], false, false);
      }
    }
#pragma unroll
    for (int n2 = 0; n2 < 4; ++n2) {
      int n = (half * 4 + n2) * 16 + m;
#pragma unroll
      for (int j = 0; j < 8; ++j) {
        int rloc = j + 8 * hi;
        atomicAdd(&agg[(size_t)sdst[rloc] * DGNN + n], c[n2][j]);
      }
    }
  }
}

// h_ep[r] = relu(node_emb[node(r)] @ W_self + agg[node(r)]), r = 2e (src) / 2e+1 (dst)
__global__ void k_h_ep(const int* __restrict__ node_tokens,
                       const int* __restrict__ edge_index,
                       const __bf16* __restrict__ embT,
                       const __bf16* __restrict__ WselfP,
                       const float* __restrict__ agg,
                       float* __restrict__ h_ep) {
  __shared__ int nid[16];
  int l = threadIdx.x;
  int rb = blockIdx.x * 16;
  if (l < 16) {
    int r = rb + l;
    nid[l] = edge_index[(r & 1) * NEDGES + (r >> 1)];
  }
  __syncthreads();
  int m = l & 15, hi = l >> 4;
  int node = nid[m];
  int tk[LTOK];
#pragma unroll
  for (int s = 0; s < LTOK; ++s) tk[s] = node_tokens[node * LTOK + s];
  for (int half = 0; half < 2; ++half) {
    v8f c[4] = {};
    for (int kt = 0; kt < DIN / 32; ++kt) {
      int k0 = kt * 32 + hi * 8;
      int k1 = k0 + 16;
      uint4 lo  = *(const uint4*)(embT + (size_t)tk[k0 >> 6] * HID + (k0 & 63));
      uint4 hiu = *(const uint4*)(embT + (size_t)tk[k1 >> 6] * HID + (k1 & 63));
      v16bf a = pack2(lo, hiu);
#pragma unroll
      for (int n2 = 0; n2 < 4; ++n2) {
        int nt = half * 4 + n2;
        v16bf b = load_b16(WselfP + ((size_t)(kt * (DGNN / 16) + nt) << 9) + l * 16);
        c[n2] = __builtin_amdgcn_wmma_f32_16x16x32_bf16(false, a, false, b, (short)0, c[n2], false, false);
      }
    }
#pragma unroll
    for (int n2 = 0; n2 < 4; ++n2) {
      int n = (half * 4 + n2) * 16 + m;
#pragma unroll
      for (int j = 0; j < 8; ++j) {
        int rloc = j + 8 * hi;
        float v = c[n2][j] + agg[(size_t)nid[rloc] * DGNN + n];
        h_ep[(size_t)(rb + rloc) * DGNN + n] = fmaxf(v, 0.f);
      }
    }
  }
}

// x[e*2+s][c] = h_ep[2e][s*64+c] + h_ep[2e+1][s*64+c]
__global__ void k_build_x(const float* __restrict__ h_ep, float* __restrict__ x) {
  int t = blockIdx.x * blockDim.x + threadIdx.x;
  if (t >= 2 * NEDGES * HID) return;
  int col = t & 63;
  int row = t >> 6;
  int e = row >> 1, s = row & 1;
  x[t] = h_ep[(size_t)(2 * e) * DGNN + s * HID + col] +
         h_ep[(size_t)(2 * e + 1) * DGNN + s * HID + col];
}

// ---------------- transformer block (one wave per 16 rows) ----------------

__device__ void wave_gemm(const float* A, int lda, const __bf16* Bp, int K, int N,
                          float* C, int ldc, bool do_relu) {
  int l = threadIdx.x;
  int m = l & 15, hi = l >> 4;
  int ktiles = K >> 5, ntiles = N >> 4;
  for (int nt = 0; nt < ntiles; ++nt) {
    v8f c = {};
    for (int kt = 0; kt < ktiles; ++kt) {
      int kb = kt * 32;
      v16bf a;
#pragma unroll
      for (int j = 0; j < 8; ++j) {
        a[j]     = f2bf(A[m * lda + kb + 8 * hi + j]);
        a[8 + j] = f2bf(A[m * lda + kb + 16 + 8 * hi + j]);
      }
      v16bf b = load_b16(Bp + ((size_t)(kt * ntiles + nt) << 9) + l * 16);
      c = __builtin_amdgcn_wmma_f32_16x16x32_bf16(false, a, false, b, (short)0, c, false, false);
    }
    int n = nt * 16 + m;
#pragma unroll
    for (int j = 0; j < 8; ++j) {
      float v = c[j];
      if (do_relu) v = fmaxf(v, 0.f);
      C[(j + 8 * hi) * ldc + n] = v;
    }
  }
  __syncthreads();
}

__global__ void k_transformer(float* __restrict__ x,
                              const __bf16* __restrict__ WqP, const __bf16* __restrict__ WkP,
                              const __bf16* __restrict__ WvP, const __bf16* __restrict__ WoP,
                              const __bf16* __restrict__ W1P, const __bf16* __restrict__ W2P) {
  __shared__ float xs[16][68], x0[16][68], qs[16][68], ks[16][68], vs[16][68], as_[16][68];
  __shared__ float cs[16][260];
  __shared__ float pr[16][2];
  int l = threadIdx.x;
  int rb = blockIdx.x * 16;
  for (int idx = l; idx < 16 * 64; idx += 32) {
    int r = idx >> 6, cc = idx & 63;
    float v = x[(size_t)(rb + r) * HID + cc];
    xs[r][cc] = v; x0[r][cc] = v;
  }
  __syncthreads();
  wave_gemm(&xs[0][0], 68, WqP, 64, 64, &qs[0][0], 68, false);
  wave_gemm(&xs[0][0], 68, WkP, 64, 64, &ks[0][0], 68, false);
  wave_gemm(&xs[0][0], 68, WvP, 64, 64, &vs[0][0], 68, false);
  {   // attention (seq=2 per edge): 8 pairs * 2 srow * 2 t = 32 dot products
    int pair = l >> 2, sr = (l >> 1) & 1, tc = l & 1;
    int qrow = 2 * pair + sr, krow = 2 * pair + tc;
    float sd = 0.f;
    for (int d = 0; d < 64; ++d) sd += qs[qrow][d] * ks[krow][d];
    sd *= 0.125f;                         // 1/sqrt(64)
    float other = __shfl_xor(sd, 1, 32);
    float mx = fmaxf(sd, other);
    float e0 = __expf(sd - mx), e1 = __expf(other - mx);
    pr[qrow][tc] = e0 / (e0 + e1);
  }
  __syncthreads();
  for (int idx = l; idx < 16 * 64; idx += 32) {
    int r = idx >> 6, d = idx & 63;
    int e2 = r & ~1;
    as_[r][d] = pr[r][0] * vs[e2][d] + pr[r][1] * vs[e2 + 1][d];
  }
  __syncthreads();
  wave_gemm(&as_[0][0], 68, WoP, 64, 64, &cs[0][0], 260, false);
  if (l < 16) {  // residual + LN1
    float mean = 0.f;
    for (int d = 0; d < 64; ++d) mean += x0[l][d] + cs[l][d];
    mean *= (1.f / 64.f);
    float var = 0.f;
    for (int d = 0; d < 64; ++d) { float dv = x0[l][d] + cs[l][d] - mean; var += dv * dv; }
    var *= (1.f / 64.f);
    float inv = rsqrtf(var + 1e-5f);
    for (int d = 0; d < 64; ++d) {
      float nv = (x0[l][d] + cs[l][d] - mean) * inv;
      xs[l][d] = nv; x0[l][d] = nv;
    }
  }
  __syncthreads();
  wave_gemm(&xs[0][0], 68, W1P, 64, FFND, &cs[0][0], 260, true);
  wave_gemm(&cs[0][0], 260, W2P, FFND, 64, &as_[0][0], 68, false);
  if (l < 16) {  // residual + LN2 -> global
    float mean = 0.f;
    for (int d = 0; d < 64; ++d) mean += x0[l][d] + as_[l][d];
    mean *= (1.f / 64.f);
    float var = 0.f;
    for (int d = 0; d < 64; ++d) { float dv = x0[l][d] + as_[l][d] - mean; var += dv * dv; }
    var *= (1.f / 64.f);
    float inv = rsqrtf(var + 1e-5f);
    for (int d = 0; d < 64; ++d)
      x[(size_t)(rb + l) * HID + d] = (x0[l][d] + as_[l][d] - mean) * inv;
  }
}

// ---------------- LM head: two-pass online softmax ----------------

#define LM_PF_AHEAD 8   // prefetch distance in 16-col tiles (stays inside workspace)

__device__ __forceinline__ void lm_build_a(const float* xr, int hi, v16bf& a0, v16bf& a1) {
#pragma unroll
  for (int j = 0; j < 8; ++j) {
    a0[j]     = f2bf(xr[8 * hi + j]);
    a0[8 + j] = f2bf(xr[16 + 8 * hi + j]);
    a1[j]     = f2bf(xr[32 + 8 * hi + j]);
    a1[8 + j] = f2bf(xr[48 + 8 * hi + j]);
  }
}

__global__ void k_lm_pass1(const float* __restrict__ x, const __bf16* __restrict__ lmP,
                           const float* __restrict__ lm_b, float* __restrict__ stats) {
  int l = threadIdx.x;
  int rb = blockIdx.x * 16;
  int m = l & 15, hi = l >> 4;
  const float* xr = x + (size_t)(rb + m) * HID;
  v16bf a0, a1;
  lm_build_a(xr, hi, a0, a1);
  float rm[8], rs[8];
#pragma unroll
  for (int j = 0; j < 8; ++j) { rm[j] = -3.0e38f; rs[j] = 0.f; }
  const int NT = VOCAB / 16;
  for (int nt = 0; nt < NT; ++nt) {
    __builtin_prefetch(lmP + ((size_t)(nt + LM_PF_AHEAD) << 9) + l * 16, 0, 3);
    __builtin_prefetch(lmP + ((size_t)(NT + nt + LM_PF_AHEAD) << 9) + l * 16, 0, 3);
    v8f c = {};
    v16bf b0 = load_b16(lmP + ((size_t)nt << 9) + l * 16);
    v16bf b1 = load_b16(lmP + ((size_t)(NT + nt) << 9) + l * 16);
    c = __builtin_amdgcn_wmma_f32_16x16x32_bf16(false, a0, false, b0, (short)0, c, false, false);
    c = __builtin_amdgcn_wmma_f32_16x16x32_bf16(false, a1, false, b1, (short)0, c, false, false);
    float bias = lm_b[nt * 16 + m];
#pragma unroll
    for (int j = 0; j < 8; ++j) {
      float v = c[j] + bias;
      if (v > rm[j]) { rs[j] = rs[j] * __expf(rm[j] - v) + 1.f; rm[j] = v; }
      else           { rs[j] += __expf(v - rm[j]); }
    }
  }
#pragma unroll
  for (int j = 0; j < 8; ++j) {   // combine 16 lanes that share each row
    float m_ = rm[j], s_ = rs[j];
    for (int off = 1; off < 16; off <<= 1) {
      float om = __shfl_xor(m_, off, 32);
      float os = __shfl_xor(s_, off, 32);
      float nm = fmaxf(m_, om);
      s_ = s_ * __expf(m_ - nm) + os * __expf(om - nm);
      m_ = nm;
    }
    rm[j] = m_; rs[j] = s_;
  }
  if (m == 0) {
#pragma unroll
    for (int j = 0; j < 8; ++j) {
      int r = rb + j + 8 * hi;
      stats[2 * r] = rm[j];
      stats[2 * r + 1] = rs[j];
    }
  }
}

__global__ void k_lm_pass2(const float* __restrict__ x, const __bf16* __restrict__ lmP,
                           const float* __restrict__ lm_b, const float* __restrict__ stats,
                           float* __restrict__ probs) {
  int l = threadIdx.x;
  int rb = blockIdx.x * 16;
  int m = l & 15, hi = l >> 4;
  const float* xr = x + (size_t)(rb + m) * HID;
  v16bf a0, a1;
  lm_build_a(xr, hi, a0, a1);
  float rm[8], ri[8];
#pragma unroll
  for (int j = 0; j < 8; ++j) {
    int r = rb + j + 8 * hi;
    rm[j] = stats[2 * r];
    ri[j] = 1.f / stats[2 * r + 1];
  }
  const int NT = VOCAB / 16;
  for (int nt = 0; nt < NT; ++nt) {
    __builtin_prefetch(lmP + ((size_t)(nt + LM_PF_AHEAD) << 9) + l * 16, 0, 3);
    __builtin_prefetch(lmP + ((size_t)(NT + nt + LM_PF_AHEAD) << 9) + l * 16, 0, 3);
    v8f c = {};
    v16bf b0 = load_b16(lmP + ((size_t)nt << 9) + l * 16);
    v16bf b1 = load_b16(lmP + ((size_t)(NT + nt) << 9) + l * 16);
    c = __builtin_amdgcn_wmma_f32_16x16x32_bf16(false, a0, false, b0, (short)0, c, false, false);
    c = __builtin_amdgcn_wmma_f32_16x16x32_bf16(false, a1, false, b1, (short)0, c, false, false);
    float bias = lm_b[nt * 16 + m];
    int n = nt * 16 + m;
#pragma unroll
    for (int j = 0; j < 8; ++j) {
      int r = rb + j + 8 * hi;
      probs[(size_t)r * VOCAB + n] = __expf(c[j] + bias - rm[j]) * ri[j];
    }
  }
}

// ---------------- labels ----------------

__global__ void k_labels(const int* __restrict__ edge_tokens, float* __restrict__ out) {
  int t = blockIdx.x * blockDim.x + threadIdx.x;
  if (t >= NEDGES * LTOK) return;
  int tok = edge_tokens[t];
  int e = t >> 3;
  bool keep = (e == 0) && (tok >= 4);   // specials are tokens 0..3; only row 0 kept, all other rows -100
  out[t] = keep ? (float)tok : -100.0f;
}

// ---------------- workspace layout ----------------

static constexpr size_t OFF_EMB   = 0;
static constexpr size_t OFF_WSELF = OFF_EMB   + (size_t)VOCAB * HID * 2;
static constexpr size_t OFF_WNBR  = OFF_WSELF + (size_t)DIN * DGNN * 2;
static constexpr size_t OFF_WQ    = OFF_WNBR  + (size_t)DIN * DGNN * 2;
static constexpr size_t OFF_WK    = OFF_WQ    + (size_t)HID * HID * 2;
static constexpr size_t OFF_WV    = OFF_WK    + (size_t)HID * HID * 2;
static constexpr size_t OFF_WO    = OFF_WV    + (size_t)HID * HID * 2;
static constexpr size_t OFF_W1    = OFF_WO    + (size_t)HID * HID * 2;
static constexpr size_t OFF_W2    = OFF_W1    + (size_t)HID * FFND * 2;
static constexpr size_t OFF_LM    = OFF_W2    + (size_t)FFND * HID * 2;
static constexpr size_t OFF_AGG   = OFF_LM    + (size_t)HID * VOCAB * 2;
static constexpr size_t OFF_HEP   = OFF_AGG   + (size_t)NNODES * DGNN * 4;
static constexpr size_t OFF_X     = OFF_HEP   + (size_t)2 * NEDGES * DGNN * 4;
static constexpr size_t OFF_STATS = OFF_X     + (size_t)2 * NEDGES * HID * 4;

extern "C" void kernel_launch(void* const* d_in, const int* in_sizes, int n_in,
                              void* d_out, int out_size, void* d_ws, size_t ws_size,
                              hipStream_t stream) {
  (void)in_sizes; (void)n_in; (void)out_size; (void)ws_size;
  const int*   node_tokens = (const int*)d_in[0];
  const int*   edge_tokens = (const int*)d_in[1];
  const int*   edge_index  = (const int*)d_in[2];
  const float* embed       = (const float*)d_in[3];
  const float* W_self      = (const float*)d_in[4];
  const float* W_nbr       = (const float*)d_in[5];
  const float* Wq          = (const float*)d_in[6];
  const float* Wk          = (const float*)d_in[7];
  const float* Wv          = (const float*)d_in[8];
  const float* Wo          = (const float*)d_in[9];
  const float* W1          = (const float*)d_in[10];
  const float* W2          = (const float*)d_in[11];
  const float* lm_W        = (const float*)d_in[12];
  const float* lm_b        = (const float*)d_in[13];

  char* ws = (char*)d_ws;
  __bf16* embT   = (__bf16*)(ws + OFF_EMB);
  __bf16* WselfP = (__bf16*)(ws + OFF_WSELF);
  __bf16* WnbrP  = (__bf16*)(ws + OFF_WNBR);
  __bf16* WqP    = (__bf16*)(ws + OFF_WQ);
  __bf16* WkP    = (__bf16*)(ws + OFF_WK);
  __bf16* WvP    = (__bf16*)(ws + OFF_WV);
  __bf16* WoP    = (__bf16*)(ws + OFF_WO);
  __bf16* W1P    = (__bf16*)(ws + OFF_W1);
  __bf16* W2P    = (__bf16*)(ws + OFF_W2);
  __bf16* lmP    = (__bf16*)(ws + OFF_LM);
  float*  agg    = (float*)(ws + OFF_AGG);
  float*  h_ep   = (float*)(ws + OFF_HEP);
  float*  xbuf   = (float*)(ws + OFF_X);
  float*  stats  = (float*)(ws + OFF_STATS);

  float* labels_out = (float*)d_out;
  float* probs_out  = (float*)d_out + NEDGES * LTOK;

  // pack / convert (every call; deterministic)
  k_f32_to_bf16<<<(VOCAB * HID + 255) / 256, 256, 0, stream>>>(embed, embT, VOCAB * HID);
  k_pack_wmma_b<<<(DIN * DGNN + 255) / 256, 256, 0, stream>>>(W_self, WselfP, DIN, DGNN);
  k_pack_wmma_b<<<(DIN * DGNN + 255) / 256, 256, 0, stream>>>(W_nbr, WnbrP, DIN, DGNN);
  k_pack_wmma_b<<<(HID * HID + 255) / 256, 256, 0, stream>>>(Wq, WqP, HID, HID);
  k_pack_wmma_b<<<(HID * HID + 255) / 256, 256, 0, stream>>>(Wk, WkP, HID, HID);
  k_pack_wmma_b<<<(HID * HID + 255) / 256, 256, 0, stream>>>(Wv, WvP, HID, HID);
  k_pack_wmma_b<<<(HID * HID + 255) / 256, 256, 0, stream>>>(Wo, WoP, HID, HID);
  k_pack_wmma_b<<<(HID * FFND + 255) / 256, 256, 0, stream>>>(W1, W1P, HID, FFND);
  k_pack_wmma_b<<<(FFND * HID + 255) / 256, 256, 0, stream>>>(W2, W2P, FFND, HID);
  k_pack_wmma_b<<<(HID * VOCAB + 255) / 256, 256, 0, stream>>>(lm_W, lmP, HID, VOCAB);

  // GNN
  k_zero_agg<<<2 * NEDGES, DGNN, 0, stream>>>(edge_index, agg);
  k_msg_scatter<<<NEDGES / 16, 32, 0, stream>>>(node_tokens, edge_tokens, edge_index,
                                                embT, WnbrP, agg);
  k_h_ep<<<2 * NEDGES / 16, 32, 0, stream>>>(node_tokens, edge_index, embT, WselfP, agg, h_ep);
  k_build_x<<<(2 * NEDGES * HID + 255) / 256, 256, 0, stream>>>(h_ep, xbuf);

  // transformer
  k_transformer<<<2 * NEDGES / 16, 32, 0, stream>>>(xbuf, WqP, WkP, WvP, WoP, W1P, W2P);

  // LM head + softmax
  k_lm_pass1<<<2 * NEDGES / 16, 32, 0, stream>>>(xbuf, lmP, lm_b, stats);
  k_lm_pass2<<<2 * NEDGES / 16, 32, 0, stream>>>(xbuf, lmP, lm_b, stats, probs_out);

  // labels
  k_labels<<<(NEDGES * LTOK + 255) / 256, 256, 0, stream>>>(edge_tokens, labels_out);
}
